// GCN_risk_13391708028988
// MI455X (gfx1250) — matrix operand order repaired and checked by reading, not compiled, not run on hardware
//
#include <hip/hip_runtime.h>

// ---------------------------------------------------------------------------
// GCN_risk for MI455X (gfx1250): bf16 WMMA GEMMs (LDS-staged B fragments,
// float4 A loads, compile-time K/tile counts) + scalar graph ops.
// ---------------------------------------------------------------------------

#define NNODES 32768
#define HDIM   64
#define DINF   32
#define NEDGES 262144
#define BATCH  8
#define NPER   4096
#define EPER   32768
#define BN_EPS 1e-5f

typedef __attribute__((ext_vector_type(16))) __bf16 v16bf;
typedef __attribute__((ext_vector_type(8)))  float  v8f;

// ---------------------------------------------------------------------------
// WMMA GEMM, O[nrows x ldo] = act( A[rowidx] (nrows x KSTEPS*32, lda=KSTEPS*32
// or larger) * W (Kact x Ho, ldW) + bias ), optional accumulate into O.
//  * KSTEPS: K / 32 (A must be zero-padded to KSTEPS*32 columns)
//  * NTILE : ceil(Ho/16); ldo == NTILE*16
//  * nrows must be a multiple of 128 (grid = nrows/128, 8 waves/block)
// W is staged once per block into LDS in WMMA B-fragment order (bf16), so the
// hot loop is: 4x global_load_b128 (A) + 2x ds_load_b128 (B) per WMMA tile.
// All guards are load/store-side only; EXEC is all ones at each WMMA.
// ---------------------------------------------------------------------------
template<int KSTEPS, int NTILE>
__global__ __launch_bounds__(256)
void k_gemm(const float* __restrict__ A, const int* __restrict__ rowidx,
            const float* __restrict__ W, int Kact, int lda,
            int Ho, int ldW, int ldo, const float* __restrict__ bias,
            int act, int accum, float* __restrict__ O)
{
    // ---- stage W into LDS, pre-swizzled as bf16 B fragments ----
    __shared__ v16bf Bf[NTILE * KSTEPS * 32];
    for (int f = threadIdx.x; f < NTILE * KSTEPS * 512; f += 256) {
        const int i    = f & 15;
        const int lane = (f >> 4) & 31;
        const int rest = f >> 9;
        const int ks   = rest % KSTEPS;
        const int t    = rest / KSTEPS;
        const int k    = ks * 32 + (lane >> 4) * 16 + i;
        const int c    = t * 16 + (lane & 15);
        const float v  = (k < Kact && c < Ho) ? W[(long)k * ldW + c] : 0.f;
        Bf[(t * KSTEPS + ks) * 32 + lane][i] = (__bf16)v;
    }
    __syncthreads();

    const int lane = threadIdx.x & 31;
    const int wave = threadIdx.x >> 5;
    const int row0 = (blockIdx.x * 8 + wave) * 16;
    const int m    = lane & 15;           // A row / B,C col within tile
    const int kh   = lane >> 4;           // lane-half selector

    const int row  = row0 + m;
    const long arow = (long)(rowidx ? rowidx[row] : row) * lda;
    const float4* A4 = (const float4*)(A + arow);

    // ---- A fragments: loaded once, reused across all NTILE column tiles ----
    v16bf afrag[KSTEPS];
#pragma unroll
    for (int ks = 0; ks < KSTEPS; ++ks) {
        const float4 p0 = A4[ks * 8 + kh * 4];
        const float4 p1 = A4[ks * 8 + kh * 4 + 1];
        const float4 p2 = A4[ks * 8 + kh * 4 + 2];
        const float4 p3 = A4[ks * 8 + kh * 4 + 3];
        afrag[ks][0]  = (__bf16)p0.x; afrag[ks][1]  = (__bf16)p0.y;
        afrag[ks][2]  = (__bf16)p0.z; afrag[ks][3]  = (__bf16)p0.w;
        afrag[ks][4]  = (__bf16)p1.x; afrag[ks][5]  = (__bf16)p1.y;
        afrag[ks][6]  = (__bf16)p1.z; afrag[ks][7]  = (__bf16)p1.w;
        afrag[ks][8]  = (__bf16)p2.x; afrag[ks][9]  = (__bf16)p2.y;
        afrag[ks][10] = (__bf16)p2.z; afrag[ks][11] = (__bf16)p2.w;
        afrag[ks][12] = (__bf16)p3.x; afrag[ks][13] = (__bf16)p3.y;
        afrag[ks][14] = (__bf16)p3.z; afrag[ks][15] = (__bf16)p3.w;
    }

#pragma unroll
    for (int t = 0; t < NTILE; ++t) {
        const int c = t * 16 + m;
        v8f acc = {0.f, 0.f, 0.f, 0.f, 0.f, 0.f, 0.f, 0.f};
        if (accum) {
#pragma unroll
            for (int r = 0; r < 8; ++r)
                if (c < Ho) acc[r] = O[(long)(row0 + r + kh * 8) * ldo + c];
        }
#pragma unroll
        for (int ks = 0; ks < KSTEPS; ++ks) {
            const v16bf b = Bf[(t * KSTEPS + ks) * 32 + lane];
            acc = __builtin_amdgcn_wmma_f32_16x16x32_bf16(
                false, afrag[ks], false, b, (short)0, acc, false, false);
        }
        const float bb = (bias && c < Ho) ? bias[c] : 0.f;
#pragma unroll
        for (int r = 0; r < 8; ++r) {
            float v = (c < Ho) ? (acc[r] + bb) : 0.f;   // zero pad columns
            if (act == 1) v = fmaxf(v, 0.f);
            O[(long)(row0 + r + kh * 8) * ldo + c] = v;
        }
    }
}

// ---------------------------------------------------------------------------
// Per-column batch stats: mean + rsqrt(var+eps). One block per column.
// ---------------------------------------------------------------------------
__global__ __launch_bounds__(256)
void k_colstats(const float* __restrict__ X, int rows, int ld, int cols,
                float* __restrict__ mean, float* __restrict__ rstd)
{
    const int c = blockIdx.x;
    if (c >= cols) return;
    float s = 0.f, s2 = 0.f;
    for (int r = threadIdx.x; r < rows; r += 256) {
        const float v = X[(long)r * ld + c];
        s += v; s2 += v * v;
    }
    __shared__ float sh[256], sh2[256];
    sh[threadIdx.x] = s; sh2[threadIdx.x] = s2;
    __syncthreads();
    for (int o = 128; o > 0; o >>= 1) {
        if (threadIdx.x < o) {
            sh[threadIdx.x]  += sh[threadIdx.x + o];
            sh2[threadIdx.x] += sh2[threadIdx.x + o];
        }
        __syncthreads();
    }
    if (threadIdx.x == 0) {
        const float mn  = sh[0] / rows;
        const float var = sh2[0] / rows - mn * mn;
        mean[c] = mn;
        rstd[c] = rsqrtf(var + BN_EPS);
    }
}

// BN apply; writes ZERO into pad columns [cols, ldo) so downstream GEMMs can
// read K padded to a multiple of 32 without guards.
__global__ __launch_bounds__(256)
void k_bn_apply(const float* __restrict__ X, int rows, int ld, int cols,
                const float* __restrict__ mean, const float* __restrict__ rstd,
                const float* __restrict__ g, const float* __restrict__ bt,
                int relu_after, float* __restrict__ O, int ldo)
{
    const long i = (long)blockIdx.x * 256 + threadIdx.x;
    if (i >= (long)rows * ldo) return;
    const int r = (int)(i / ldo), c = (int)(i % ldo);
    float v = 0.f;
    if (c < cols) {
        v = g[c] * (X[(long)r * ld + c] - mean[c]) * rstd[c] + bt[c];
        if (relu_after) v = fmaxf(v, 0.f);
    }
    O[(long)r * ldo + c] = v;
}

// GCN self-loop + bias (+ optional residual): O = [x +] h*dinv2[row] + b[col]
__global__ __launch_bounds__(256)
void k_selfinit(const float* __restrict__ x, const float* __restrict__ h,
                const float* __restrict__ dinv2, const float* __restrict__ bias,
                int rows, int cols, float* __restrict__ O)
{
    const long i = (long)blockIdx.x * 256 + threadIdx.x;
    if (i >= (long)rows * cols) return;
    const int r = (int)(i / cols), c = (int)(i % cols);
    float v = h[i] * dinv2[r] + bias[c];
    if (x) v += x[i];
    O[i] = v;
}

// Edge scatter: O[dst] += h[src] * dinv[src]*dinv[dst]   (fp32 atomics)
__global__ __launch_bounds__(256)
void k_scatter(const float* __restrict__ h, const int* __restrict__ src,
               const int* __restrict__ dst, const float* __restrict__ dinv,
               int ne, int cols, float* __restrict__ O)
{
    const long i = (long)blockIdx.x * 256 + threadIdx.x;
    if (i >= (long)ne * cols) return;
    const int e = (int)(i / cols), c = (int)(i % cols);
    const int s = src[e], d = dst[e];
    atomicAdd(&O[(long)d * cols + c], h[(long)s * cols + c] * dinv[s] * dinv[d]);
}

// Small dense FC: one thread per output. act: 0 none, 1 relu, 2 sigmoid.
__global__ __launch_bounds__(256)
void k_fc(const float* __restrict__ X, int rows, int K, int ld,
          const float* __restrict__ W, int Ho, const float* __restrict__ bias,
          int act, float* __restrict__ O)
{
    const long i = (long)blockIdx.x * 256 + threadIdx.x;
    if (i >= (long)rows * Ho) return;
    const int r = (int)(i / Ho), o = (int)(i % Ho);
    float s = bias ? bias[o] : 0.f;
    for (int k = 0; k < K; ++k) s += X[(long)r * ld + k] * W[(long)k * Ho + o];
    if (act == 1) s = fmaxf(s, 0.f);
    else if (act == 2) s = 1.f / (1.f + __expf(-s));
    O[i] = s;
}

__global__ __launch_bounds__(256)
void k_softmax(const float* __restrict__ X, int L, float* __restrict__ O)
{
    __shared__ float sh[256];
    const float* x = X + (long)blockIdx.x * L;
    float mx = -3.4e38f;
    for (int i = threadIdx.x; i < L; i += 256) mx = fmaxf(mx, x[i]);
    sh[threadIdx.x] = mx; __syncthreads();
    for (int o = 128; o > 0; o >>= 1) {
        if (threadIdx.x < o) sh[threadIdx.x] = fmaxf(sh[threadIdx.x], sh[threadIdx.x + o]);
        __syncthreads();
    }
    mx = sh[0]; __syncthreads();
    float s = 0.f;
    for (int i = threadIdx.x; i < L; i += 256) s += __expf(x[i] - mx);
    sh[threadIdx.x] = s; __syncthreads();
    for (int o = 128; o > 0; o >>= 1) {
        if (threadIdx.x < o) sh[threadIdx.x] += sh[threadIdx.x + o];
        __syncthreads();
    }
    const float inv = 1.f / sh[0];
    float* op = O + (long)blockIdx.x * L;
    for (int i = threadIdx.x; i < L; i += 256) op[i] = __expf(x[i] - mx) * inv;
}

__global__ __launch_bounds__(256)
void k_fill(float* p, int n, float v)
{ const int i = blockIdx.x * 256 + threadIdx.x; if (i < n) p[i] = v; }

__global__ __launch_bounds__(256)
void k_deg(const int* __restrict__ dst, int ne, float* __restrict__ deg)
{ const int i = blockIdx.x * 256 + threadIdx.x; if (i < ne) atomicAdd(&deg[dst[i]], 1.f); }

__global__ __launch_bounds__(256)
void k_dinv(const float* __restrict__ deg, float* __restrict__ dinv,
            float* __restrict__ dinv2, int n)
{
    const int i = blockIdx.x * 256 + threadIdx.x;
    if (i < n) { const float d = rsqrtf(deg[i]); dinv[i] = d; dinv2[i] = d * d; }
}

// ---------------------------------------------------------------------------
extern "C" void kernel_launch(void* const* d_in, const int* in_sizes, int n_in,
                              void* d_out, int out_size, void* d_ws, size_t ws_size,
                              hipStream_t stream)
{
    (void)in_sizes; (void)out_size; (void)ws_size;
    auto F = [&](int i) { return (const float*)d_in[i]; };

    // ---- params: dict insertion order, leaves flattened ----
    struct Res { const float *W, *b, *g, *bt; };
    int pi = 2;                                   // 0:x 1:global_x
    const float* xin = F(0);
    const float *ciW = F(pi++), *cib = F(pi++);
    auto load_res = [&](Res* a, int n) {
        for (int i = 0; i < n; ++i) { a[i].W = F(pi++); a[i].b = F(pi++);
                                      a[i].g = F(pi++); a[i].bt = F(pi++); } };
    Res deep[4];   load_res(deep, 4);
    Res pickb[3];  load_res(pickb, 3);
    const float *poW = F(pi++), *pob = F(pi++);
    const float *pfW0 = F(pi++), *pfb0 = F(pi++), *pfW1 = F(pi++), *pfb1 = F(pi++);
    Res placeb[3]; load_res(placeb, 3);
    const float *plW = F(pi++), *plb = F(pi++);
    const float *plfW0 = F(pi++), *plfb0 = F(pi++), *plfW1 = F(pi++), *plfb1 = F(pi++);
    Res attb[4];   load_res(attb, 4);
    const float* ae[6];  for (int i = 0; i < 6; ++i) ae[i] = F(pi++);  // lins W,b x3
    const float* aebn[4];for (int i = 0; i < 4; ++i) aebn[i] = F(pi++);// bns g,bt x2
    const float *afW0 = F(pi++), *afb0 = F(pi++), *afW1 = F(pi++), *afb1 = F(pi++);
    Res fortb[4];  load_res(fortb, 4);
    const float* fe[6];  for (int i = 0; i < 6; ++i) fe[i] = F(pi++);
    const float* febn[4];for (int i = 0; i < 4; ++i) febn[i] = F(pi++);
    const float *ffW0 = F(pi++), *ffb0 = F(pi++), *ffW1 = F(pi++), *ffb1 = F(pi++);
    Res valb[4];   load_res(valb, 4);
    const float *v1W = F(pi++), *v1b = F(pi++), *v2W = F(pi++), *v2b = F(pi++);
    const int* ei  = (const int*)d_in[n_in - 1];
    const int* src = ei;
    const int* dst = ei + NEDGES;

    // ---- workspace carve-up (fp32) ----
    float* w = (float*)d_ws; size_t off = 0;
    auto alloc = [&](size_t n) { float* p = w + off; off += n; return p; };
    float* DEG   = alloc(NNODES);
    float* DINV  = alloc(NNODES);
    float* DINV2 = alloc(NNODES);
    float* MEAN  = alloc(128);
    float* RSTD  = alloc(128);
    float* HMAIN = alloc((size_t)NNODES * HDIM);
    float* BUFA  = alloc((size_t)NNODES * HDIM);
    float* BUFT  = alloc((size_t)NNODES * HDIM);
    float* BUFG  = alloc((size_t)NNODES * HDIM);
    float* SCG   = alloc(NNODES);
    float* SCO   = alloc(NNODES);
    float* FCT   = alloc(BATCH * 64);
    float* LOGIT = alloc((size_t)BATCH * (EPER + 1));
    float* E1    = alloc((size_t)NEDGES * 32);
    float* E2    = alloc((size_t)NEDGES * 32);
    float* ESC   = alloc(NEDGES);

    float* out         = (float*)d_out;
    float* out_pick    = out;
    float* out_place   = out + BATCH * NPER;
    float* out_attack  = out + 2 * BATCH * NPER;
    float* out_fortify = out_attack + BATCH * (EPER + 1);
    float* out_value   = out_fortify + BATCH * (EPER + 1);

    const int gNE  = (NNODES * HDIM) / 256;   // node elementwise
    const int gSC  = (NEDGES * HDIM) / 256;   // edge scatter, 64ch
    const int gGn  = NNODES / 128;            // node GEMM blocks
    const int gGe  = NEDGES / 128;            // edge GEMM blocks

    // ---- degree / normalization ----
    k_fill<<<NNODES / 256, 256, 0, stream>>>(DEG, NNODES, 1.0f);
    k_deg <<<NEDGES / 256, 256, 0, stream>>>(dst, NEDGES, DEG);
    k_dinv<<<NNODES / 256, 256, 0, stream>>>(DEG, DINV, DINV2, NNODES);

    // ---- res+ block: O = X + GCN(relu(BN(X)))  (in-place safe) ----
    auto res_block = [&](const float* X, float* O, const Res& p) {
        k_colstats<<<HDIM, 256, 0, stream>>>(X, NNODES, HDIM, HDIM, MEAN, RSTD);
        k_bn_apply<<<gNE, 256, 0, stream>>>(X, NNODES, HDIM, HDIM, MEAN, RSTD,
                                            p.g, p.bt, 1, BUFT, HDIM);
        k_gemm<2, 4><<<gGn, 256, 0, stream>>>(BUFT, nullptr, p.W, HDIM, HDIM,
                                              HDIM, HDIM, HDIM, nullptr, 0, 0, BUFG);
        k_selfinit<<<gNE, 256, 0, stream>>>(X, BUFG, DINV2, p.b, NNODES, HDIM, O);
        k_scatter<<<gSC, 256, 0, stream>>>(BUFG, src, dst, DINV, NEDGES, HDIM, O);
    };

    // ---- trunk: conv_init + 4 deep blocks -> HMAIN ----
    k_gemm<1, 4><<<gGn, 256, 0, stream>>>(xin, nullptr, ciW, DINF, DINF,
                                          HDIM, HDIM, HDIM, nullptr, 0, 0, BUFG);
    k_selfinit<<<gNE, 256, 0, stream>>>(nullptr, BUFG, DINV2, cib, NNODES, HDIM, HMAIN);
    k_scatter<<<gSC, 256, 0, stream>>>(BUFG, src, dst, DINV, NEDGES, HDIM, HMAIN);
    for (int i = 0; i < 4; ++i) res_block(HMAIN, HMAIN, deep[i]);

    // ---- node head (pick / place) ----
    auto node_head = [&](Res* blocks, int nb, const float* oW, const float* ob,
                         const float* fW0, const float* fb0,
                         const float* fW1, const float* fb1, float* outp) {
        const float* X = HMAIN;
        for (int i = 0; i < nb; ++i) { res_block(X, BUFA, blocks[i]); X = BUFA; }
        k_fc<<<NNODES / 256, 256, 0, stream>>>(X, NNODES, HDIM, HDIM, oW, 1,
                                               nullptr, 0, SCG);
        k_selfinit<<<NNODES / 256, 256, 0, stream>>>(nullptr, SCG, DINV2, ob,
                                                     NNODES, 1, SCO);
        k_scatter<<<NEDGES / 256, 256, 0, stream>>>(SCG, src, dst, DINV,
                                                    NEDGES, 1, SCO);
        k_fc<<<2, 256, 0, stream>>>(SCO, BATCH, NPER, NPER, fW0, 64, fb0, 1, FCT);
        k_fc<<<(BATCH * NPER) / 256, 256, 0, stream>>>(FCT, BATCH, 64, 64,
                                                       fW1, NPER, fb1, 0, LOGIT);
        k_softmax<<<BATCH, 256, 0, stream>>>(LOGIT, NPER, outp);
    };
    node_head(pickb, 3, poW, pob, pfW0, pfb0, pfW1, pfb1, out_pick);
    node_head(placeb, 3, plW, plb, plfW0, plfb0, plfW1, plfb1, out_place);

    // ---- edge head (attack / fortify) ----
    // concat(h[src],h[dst]) @ W128x28 done as two gather-GEMMs over split W.
    auto edge_head = [&](Res* blocks, const float** L, const float** BN,
                         const float* fW0, const float* fb0,
                         const float* fW1, const float* fb1, float* outp) {
        const float* X = HMAIN;
        for (int i = 0; i < 4; ++i) { res_block(X, BUFA, blocks[i]); X = BUFA; }
        k_gemm<2, 2><<<gGe, 256, 0, stream>>>(BUFA, src, L[0], HDIM, HDIM,
                                              28, 28, 32, nullptr, 0, 0, E1);
        k_gemm<2, 2><<<gGe, 256, 0, stream>>>(BUFA, dst, L[0] + 64 * 28, HDIM,
                                              HDIM, 28, 28, 32, L[1], 1, 1, E1);
        k_colstats<<<28, 256, 0, stream>>>(E1, NEDGES, 32, 28, MEAN, RSTD);
        k_bn_apply<<<(NEDGES * 32) / 256, 256, 0, stream>>>(
            E1, NEDGES, 32, 28, MEAN, RSTD, BN[0], BN[1], 0, E2, 32);
        k_gemm<1, 2><<<gGe, 256, 0, stream>>>(E2, nullptr, L[2], 28, 32,
                                              28, 28, 32, L[3], 1, 0, E1);
        k_colstats<<<28, 256, 0, stream>>>(E1, NEDGES, 32, 28, MEAN, RSTD);
        k_bn_apply<<<(NEDGES * 32) / 256, 256, 0, stream>>>(
            E1, NEDGES, 32, 28, MEAN, RSTD, BN[2], BN[3], 0, E2, 32);
        k_fc<<<NEDGES / 256, 256, 0, stream>>>(E2, NEDGES, 28, 32, L[4], 1,
                                               L[5], 2, ESC);
        k_fc<<<2, 256, 0, stream>>>(ESC, BATCH, EPER, EPER, fW0, 64, fb0, 1, FCT);
        k_fc<<<(BATCH * (EPER + 1) + 255) / 256, 256, 0, stream>>>(
            FCT, BATCH, 64, 64, fW1, EPER + 1, fb1, 0, LOGIT);
        k_softmax<<<BATCH, 256, 0, stream>>>(LOGIT, EPER + 1, outp);
    };
    edge_head(attb,  ae, aebn, afW0, afb0, afW1, afb1, out_attack);
    edge_head(fortb, fe, febn, ffW0, ffb0, ffW1, ffb1, out_fortify);

    // ---- value head ----
    {
        const float* X = HMAIN;
        for (int i = 0; i < 4; ++i) { res_block(X, BUFA, valb[i]); X = BUFA; }
        k_colstats<<<HDIM, 256, 0, stream>>>(X, NNODES, HDIM, HDIM, MEAN, RSTD);
        k_fc<<<1, 256, 0, stream>>>(MEAN, 1, HDIM, HDIM, v1W, HDIM, v1b, 1, FCT);
        k_fc<<<1, 256, 0, stream>>>(FCT, 1, HDIM, HDIM, v2W, 6, v2b, 2, out_value);
    }
}
